// BeamDecoder_60327110640464
// MI455X (gfx1250) — matrix operand order repaired and checked by reading, not compiled, not run on hardware
//
#include <hip/hip_runtime.h>
#include <hip/hip_bf16.h>
#include <math.h>

// ---------------------------------------------------------------------------
// Problem constants (from the reference)
// ---------------------------------------------------------------------------
#define NN   2048          // nodes
#define HH   1024          // hidden
#define TT   (NN + NN/8)   // 2304 decode steps
#define DEC_THREADS 256    // 8 waves (wave32)
#define NODES_PER_THREAD (NN / DEC_THREADS)   // 8

typedef __attribute__((ext_vector_type(16))) __bf16 v16bf;
typedef __attribute__((ext_vector_type(8)))  float  v8f;

union FragB16 { v16bf v; uint4 q[2]; };
union FragC32 { v8f  v; float f[8]; };

// ---------------------------------------------------------------------------
// f32 -> bf16 convert (grid-stride)
// ---------------------------------------------------------------------------
__global__ void conv_f32_bf16(const float* __restrict__ src,
                              __bf16* __restrict__ dst, int n) {
  for (int i = blockIdx.x * blockDim.x + threadIdx.x; i < n;
       i += gridDim.x * blockDim.x)
    dst[i] = (__bf16)src[i];
}

// Wq is [H, H+2] row-major. Extract the first H columns as bf16 and the
// "load" column (index H) as f32.
__global__ void conv_wq(const float* __restrict__ Wq,
                        __bf16* __restrict__ Wqbf,
                        float* __restrict__ wq_load, int H) {
  int total = H * H;
  int gid = blockIdx.x * blockDim.x + threadIdx.x;
  for (int i = gid; i < total; i += gridDim.x * blockDim.x) {
    int j = i / H, k = i - j * H;
    Wqbf[i] = (__bf16)Wq[(size_t)j * (H + 2) + k];
  }
  if (gid < H) wq_load[gid] = Wq[(size_t)gid * (H + 2) + H];
}

// ---------------------------------------------------------------------------
// WMMA bf16 NT GEMM:  C[M,N] = A[M,K] * B[N,K]^T (+ bias[col])
// A, B row-major bf16.  Each wave owns a 32x32 output tile (2x2 WMMA frags).
// Optional f32 and/or bf16 outputs.
// ---------------------------------------------------------------------------
__device__ inline v16bf load_frag_a(const __bf16* base, int ld, int row0,
                                    int k0, int lane) {
  // 16-bit A 16x32 layout: lane<16 holds K 0..7 / 16..23; lane>=16 holds
  // K 8..15 / 24..31 (ISA 7.12.2).
  int r  = row0 + (lane & 15);
  int kh = (lane >> 4) * 8;
  const __bf16* p = base + (size_t)r * ld + k0 + kh;
  FragB16 f;
  f.q[0] = *(const uint4*)(p);
  f.q[1] = *(const uint4*)(p + 16);
  return f.v;
}

__device__ inline v16bf load_frag_b(const __bf16* base, int ld, int col0,
                                    int k0, int lane) {
  // 16-bit B 32x16 layout: lane n<16 holds column n, K 0..7 / 16..23;
  // lane n+16 holds column n, K 8..15 / 24..31 (matches A).
  // For NT GEMM, B column n == row (col0+n) of the row-major operand.
  int c  = col0 + (lane & 15);
  int kh = (lane >> 4) * 8;
  const __bf16* p = base + (size_t)c * ld + k0 + kh;
  FragB16 f;
  f.q[0] = *(const uint4*)(p);
  f.q[1] = *(const uint4*)(p + 16);
  return f.v;
}

__device__ inline void store_tile(v8f c, int m0, int n0, int lane,
                                  float* outF, __bf16* outB,
                                  const float* bias, int ldc) {
  int col   = n0 + (lane & 15);
  int rbase = m0 + ((lane >> 4) << 3);
  float b = bias ? bias[col] : 0.0f;
  FragC32 fc; fc.v = c;
#pragma unroll
  for (int r = 0; r < 8; ++r) {
    float val = fc.f[r] + b;
    size_t idx = (size_t)(rbase + r) * ldc + col;
    if (outF) outF[idx] = val;
    if (outB) outB[idx] = (__bf16)val;
  }
}

__global__ void wmma_gemm_nt(const __bf16* __restrict__ A, int lda,
                             const __bf16* __restrict__ B, int ldb,
                             int M, int N, int Kd,
                             float* outF, __bf16* outB,
                             const float* bias, int ldc) {
  int lane = threadIdx.x & 31;
  int wave = (blockIdx.x * blockDim.x + threadIdx.x) >> 5;
  int tn = N >> 5, tm = M >> 5;
  if (wave >= tm * tn) return;
  int m0 = (wave / tn) << 5;
  int n0 = (wave % tn) << 5;

  v8f c00 = {}, c01 = {}, c10 = {}, c11 = {};
  for (int k = 0; k < Kd; k += 32) {
    v16bf a0 = load_frag_a(A, lda, m0,      k, lane);
    v16bf a1 = load_frag_a(A, lda, m0 + 16, k, lane);
    v16bf b0 = load_frag_b(B, ldb, n0,      k, lane);
    v16bf b1 = load_frag_b(B, ldb, n0 + 16, k, lane);
    c00 = __builtin_amdgcn_wmma_f32_16x16x32_bf16(false, a0, false, b0,
                                                  (short)0, c00, false, false);
    c01 = __builtin_amdgcn_wmma_f32_16x16x32_bf16(false, a0, false, b1,
                                                  (short)0, c01, false, false);
    c10 = __builtin_amdgcn_wmma_f32_16x16x32_bf16(false, a1, false, b0,
                                                  (short)0, c10, false, false);
    c11 = __builtin_amdgcn_wmma_f32_16x16x32_bf16(false, a1, false, b1,
                                                  (short)0, c11, false, false);
  }
  store_tile(c00, m0,      n0,      lane, outF, outB, bias, ldc);
  store_tile(c01, m0,      n0 + 16, lane, outF, outB, bias, ldc);
  store_tile(c10, m0 + 16, n0,      lane, outF, outB, bias, ldc);
  store_tile(c11, m0 + 16, n0 + 16, lane, outF, outB, bias, ldc);
}

// ---------------------------------------------------------------------------
// u[i] = K[i,:].wq_load ; v[i] = K[i,:].bq   (K stored as bf16)
// One 256-thread block per row.
// ---------------------------------------------------------------------------
__global__ void uv_kernel(const __bf16* __restrict__ Kbf,
                          const float* __restrict__ wql,
                          const float* __restrict__ bq,
                          float* __restrict__ u, float* __restrict__ v, int H) {
  __shared__ float su[256], sv[256];
  int row = blockIdx.x, tid = threadIdx.x;
  const __bf16* kr = Kbf + (size_t)row * H;
  float au = 0.0f, av = 0.0f;
  for (int j = tid; j < H; j += 256) {
    float kv = (float)kr[j];
    au += kv * wql[j];
    av += kv * bq[j];
  }
  su[tid] = au; sv[tid] = av;
  __syncthreads();
  for (int s = 128; s > 0; s >>= 1) {
    if (tid < s) { su[tid] += su[tid + s]; sv[tid] += sv[tid + s]; }
    __syncthreads();
  }
  if (tid == 0) { u[row] = su[0]; v[row] = sv[0]; }
}

// ---------------------------------------------------------------------------
// Serial greedy decode.  Single 256-thread (8-wave) workgroup, T steps.
//   s_i = scale * ( ST[last, i] + (load/cap)*u[i] + v[i] )
// Per step: each thread scans 8 contiguous nodes (two float4 loads from the
// L2-resident ST row), wave-level argmax via __shfl_xor butterfly (no
// barriers), then 8 wave partials merged by thread 0 -> 2 barriers/step.
// Tie-breaking prefers the smaller index at every level (jnp.argmax
// first-max semantics).  Outputs: tour (T+1 floats) then scores (T floats).
// ---------------------------------------------------------------------------
__global__ void decode_kernel(const float* __restrict__ ST,
                              const float* __restrict__ u,
                              const float* __restrict__ v,
                              const float* __restrict__ demands,
                              const int* __restrict__ cap_p,
                              const int* __restrict__ depot_p,
                              float* __restrict__ out, int T) {
  __shared__ float s_dem[NN];
  __shared__ float s_u[NN];
  __shared__ float s_v[NN];
  __shared__ unsigned char s_vis[NN];
  __shared__ float rv[DEC_THREADS / 32];
  __shared__ int   ri[DEC_THREADS / 32];
  __shared__ float sh_load;
  __shared__ int   sh_last, sh_done, sh_served;

  const int tid  = threadIdx.x;
  const int lane = tid & 31;
  const int wid  = tid >> 5;

  for (int i = tid; i < NN; i += DEC_THREADS) {
    s_dem[i] = demands[i];
    s_u[i]   = u[i];
    s_v[i]   = v[i];
    s_vis[i] = 0;
  }
  __syncthreads();
  const float capf  = (float)(*cap_p);
  const int   depot = *depot_p;
  if (tid == 0) {
    s_vis[depot] = 1;
    sh_load = capf; sh_last = depot; sh_done = 0; sh_served = 0;
    out[0] = (float)depot;
  }
  __syncthreads();

  const float scale   = 0.5f * (1.0f / 32.0f);  // ALPHA * 1/sqrt(HID)
  const float NEG_INF = -__builtin_inff();
  const int   base    = tid * NODES_PER_THREAD;

  for (int t = 0; t < T; ++t) {
    const float load = sh_load;
    const int   last = sh_last;
    const float lf   = load / capf;
    const float* srow = ST + (size_t)last * NN;

    // ---- per-thread scan of 8 contiguous nodes (two float4 loads) ----
    float bv = NEG_INF;
    int   bi = 0;
    const float4* s4 = (const float4*)(srow + base);
    float vals[NODES_PER_THREAD];
    {
      float4 a = s4[0], b = s4[1];
      vals[0] = a.x; vals[1] = a.y; vals[2] = a.z; vals[3] = a.w;
      vals[4] = b.x; vals[5] = b.y; vals[6] = b.z; vals[7] = b.w;
    }
#pragma unroll
    for (int e = 0; e < NODES_PER_THREAD; ++e) {
      int i = base + e;
      float sc = scale * (vals[e] + lf * s_u[i] + s_v[i]);
      bool feas = (s_vis[i] == 0) && (s_dem[i] <= load);
      float val = feas ? sc : NEG_INF;
      if (val > bv) { bv = val; bi = i; }   // ascending scan keeps first max
    }

    // ---- wave32 butterfly argmax (no barriers) ----
#pragma unroll
    for (int m = 16; m >= 1; m >>= 1) {
      float ov = __shfl_xor(bv, m, 32);
      int   oi = __shfl_xor(bi, m, 32);
      if (ov > bv || (ov == bv && oi < bi)) { bv = ov; bi = oi; }
    }
    if (lane == 0) { rv[wid] = bv; ri[wid] = bi; }
    __syncthreads();

    // ---- thread 0: merge 8 wave partials, update state, emit outputs ----
    if (tid == 0) {
      float best = rv[0]; int cand = ri[0];
#pragma unroll
      for (int w = 1; w < DEC_THREADS / 32; ++w) {
        float v2 = rv[w]; int i2 = ri[w];
        if (v2 > best || (v2 == best && i2 < cand)) { best = v2; cand = i2; }
      }
      bool has  = (best > NEG_INF);
      bool done = (sh_done != 0);
      bool take = has && !done;
      int  nxt  = take ? cand : depot;
      float step_score = 0.0f;
      if (take) {
        step_score = scale * (srow[cand] + lf * s_u[cand] + s_v[cand]);
        s_vis[cand] = 1;
        sh_served += 1;
        sh_load = load - s_dem[cand];
      } else {
        sh_load = capf;
        if (sh_served >= NN - 1) sh_done = 1;   // all customers visited
      }
      sh_last = nxt;
      out[1 + t]       = (float)nxt;   // tour
      out[(T + 1) + t] = step_score;   // scores
    }
    __syncthreads();
  }
}

// ---------------------------------------------------------------------------
// Launch
// ---------------------------------------------------------------------------
extern "C" void kernel_launch(void* const* d_in, const int* in_sizes, int n_in,
                              void* d_out, int out_size, void* d_ws,
                              size_t ws_size, hipStream_t stream) {
  const float* node_emb = (const float*)d_in[0];  // [N, H]
  const float* demands  = (const float*)d_in[1];  // [N]
  const float* Wq       = (const float*)d_in[2];  // [H, H+2]
  const float* bq       = (const float*)d_in[3];  // [H]
  const float* Wk       = (const float*)d_in[4];  // [H, H]
  const float* bk       = (const float*)d_in[5];  // [H]
  const int*   cap_p    = (const int*)d_in[6];
  const int*   depot_p  = (const int*)d_in[7];
  (void)in_sizes; (void)n_in; (void)out_size; (void)ws_size;

  char* ws = (char*)d_ws;
  size_t off = 0;
  auto alloc = [&](size_t bytes) -> void* {
    void* p = (void*)(ws + off);
    off = (off + bytes + 255) & ~(size_t)255;
    return p;
  };
  __bf16* Ebf  = (__bf16*)alloc((size_t)NN * HH * 2);
  __bf16* Wkbf = (__bf16*)alloc((size_t)HH * HH * 2);
  __bf16* Wqbf = (__bf16*)alloc((size_t)HH * HH * 2);
  float*  wql  = (float*) alloc((size_t)HH * 4);
  __bf16* Kbf  = (__bf16*)alloc((size_t)NN * HH * 2);
  __bf16* Qbf  = (__bf16*)alloc((size_t)NN * HH * 2);
  float*  ST   = (float*) alloc((size_t)NN * NN * 4);
  float*  u    = (float*) alloc((size_t)NN * 4);
  float*  v    = (float*) alloc((size_t)NN * 4);

  // 1) bf16 conversions
  conv_f32_bf16<<<1024, 256, 0, stream>>>(node_emb, Ebf, NN * HH);
  conv_f32_bf16<<<512, 256, 0, stream>>>(Wk, Wkbf, HH * HH);
  conv_wq<<<512, 256, 0, stream>>>(Wq, Wqbf, wql, HH);

  // 2) K = E @ Wk^T + bk   (bf16 out)
  {
    int waves = (NN / 32) * (HH / 32);
    wmma_gemm_nt<<<waves / 8, 256, 0, stream>>>(Ebf, HH, Wkbf, HH,
                                                NN, HH, HH,
                                                nullptr, Kbf, bk, HH);
  }
  // 3) Q = E @ Wq1^T       (bf16 out)
  {
    int waves = (NN / 32) * (HH / 32);
    wmma_gemm_nt<<<waves / 8, 256, 0, stream>>>(Ebf, HH, Wqbf, HH,
                                                NN, HH, HH,
                                                nullptr, Qbf, nullptr, HH);
  }
  // 4) ST = Q @ K^T        (f32 out; row j = scores when last == j)
  {
    int waves = (NN / 32) * (NN / 32);
    wmma_gemm_nt<<<waves / 8, 256, 0, stream>>>(Qbf, HH, Kbf, HH,
                                                NN, NN, HH,
                                                ST, nullptr, nullptr, NN);
  }
  // 5) u = K . wq_load ; v = K . bq
  uv_kernel<<<NN, 256, 0, stream>>>(Kbf, wql, bq, u, v, HH);

  // 6) serial greedy decode
  decode_kernel<<<1, DEC_THREADS, 0, stream>>>(ST, u, v, demands, cap_p,
                                               depot_p, (float*)d_out, TT);
}
